// MV2DFusionHeadForONNX_84499186581926
// MI455X (gfx1250) — matrix operand order repaired and checked by reading, not compiled, not run on hardware
//
#include <hip/hip_runtime.h>

// ---------------- problem constants ----------------
#define LYR 6
#define BB  8
#define QQ  2048
#define CC  256
#define PP  128
#define FFD 1024
#define MTOT (BB*QQ)        // 16384 rows
#define BM  64              // rows per block

typedef __attribute__((ext_vector_type(16))) _Float16 v16h;
typedef __attribute__((ext_vector_type(8)))  _Float16 v8h;
typedef __attribute__((ext_vector_type(4)))  _Float16 v4h;
typedef __attribute__((ext_vector_type(8)))  float    v8f;

// ---------------- Tensor Data Mover staging (gfx1250) ------------------------
#if defined(__gfx1250__) && __has_builtin(__builtin_amdgcn_tensor_load_to_lds)
#define HAVE_TDM 1
typedef unsigned int u32;
typedef __attribute__((ext_vector_type(4))) unsigned int u32x4;
typedef __attribute__((ext_vector_type(8))) int          i32x8;
typedef __attribute__((ext_vector_type(4))) int          i32x4;

__device__ __forceinline__ unsigned lds_addr_of(const void* p) {
  return (unsigned)(unsigned long long)
         (__attribute__((address_space(3))) const void*)p;
}

// 2D tile: `rows` rows of `rowElems` f16 elements, global row stride
// `gStrideElems`, LDS destination padded by (padAmt+1) DWORDs after every
// 2^(padInt+1) DWORDs (TDM pad feature -> reproduces padded LDS leading dims).
__device__ __forceinline__ void tdm_load_2d(u32 ldsAddr, const void* gptr,
                                            u32 rows, u32 rowElems,
                                            u32 gStrideElems,
                                            u32 padInt, u32 padAmt) {
  const unsigned long long ga = (unsigned long long)gptr;
  u32x4 g0;
  g0[0] = 1u;                                            // count=1 (valid D#)
  g0[1] = ldsAddr;                                       // lds_addr
  g0[2] = (u32)ga;                                       // global_addr[31:0]
  g0[3] = (u32)((ga >> 32) & 0x01FFFFFFull) | (2u << 30);// addr[56:32] | type=2
  i32x8 g1;
  g1[0] = (int)((1u << 16)            // data_size = 1 (2 bytes)
              | (1u << 20)            // pad_enable
              | (padInt << 22) | (padAmt << 25));
  g1[1] = (int)((rowElems & 0xFFFFu) << 16);             // tensor_dim0[15:0]
  g1[2] = (int)(((rowElems >> 16) & 0xFFFFu)             // tensor_dim0[31:16]
              | ((rows & 0xFFFFu) << 16));               // tensor_dim1[15:0]
  g1[3] = (int)(((rows >> 16) & 0xFFFFu)                 // tensor_dim1[31:16]
              | ((rowElems & 0xFFFFu) << 16));           // tile_dim0
  g1[4] = (int)(rows & 0xFFFFu);                         // tile_dim1 (dim2=0)
  g1[5] = (int)gStrideElems;                             // dim0_stride[31:0]
  g1[6] = 0;                                             // stride hi / dim1_stride
  g1[7] = 0;
  const i32x4 z4 = {0, 0, 0, 0};
  const i32x8 z8 = {0, 0, 0, 0, 0, 0, 0, 0};
  // amdgpu-toolchain (clang-23) 6-arg form: (g0, g1, g2, g3, g4, cpol)
  __builtin_amdgcn_tensor_load_to_lds(g0, g1, z4, z4, z8, 0);
}
#define TDM_WAIT() __builtin_amdgcn_s_wait_tensorcnt(0)
#endif

// ---------------- WMMA fragment helpers (CDNA5 §7.12.2 layouts) -------------
__device__ __forceinline__ v16h frag_a(const _Float16* s, int ld, int lane) {
  const int mrow = lane & 15;
  const int mh8  = (lane >> 4) << 3;
  const _Float16* r = s + mrow * ld + mh8;
  const v8h lo = *(const v8h*)(r);
  const v8h hi = *(const v8h*)(r + 16);
  return __builtin_shufflevector(lo, hi, 0,1,2,3,4,5,6,7,8,9,10,11,12,13,14,15);
}

__device__ __forceinline__ v16h frag_bt(const _Float16* s, int ld, int lane) {
  const int n    = lane & 15;
  const int kh16 = (lane >> 4) << 3;
  const _Float16* r = s + n * ld + kh16;
  const v8h lo = *(const v8h*)(r);
  const v8h hi = *(const v8h*)(r + 16);
  return __builtin_shufflevector(lo, hi, 0,1,2,3,4,5,6,7,8,9,10,11,12,13,14,15);
}

__device__ __forceinline__ v8f wmma32(v16h a, v16h b, v8f c) {
  return __builtin_amdgcn_wmma_f32_16x16x32_f16(false, a, false, b,
                                                (short)0, c, false, false);
}

#define VZERO8 {0.f,0.f,0.f,0.f,0.f,0.f,0.f,0.f}

// -------- weight prep: f32 -> f16, TRANSPOSED to [n][k] (B^T), once ---------
__global__ void prep_weights(const float* w1, const float* w2, const float* pw,
                             const float* posw, const float* gw, const float* bew,
                             _Float16* w1t, _Float16* w2t, _Float16* pwt,
                             _Float16* poswt, _Float16* gwt, _Float16* bewt) {
  const int t0 = blockIdx.x * blockDim.x + threadIdx.x;
  const int st = gridDim.x * blockDim.x;
  const int n1 = LYR*FFD*CC;
  for (int i = t0; i < n1; i += st) {
    const int l = i / (FFD*CC), rem = i % (FFD*CC), n = rem / CC, k = rem % CC;
    w1t[i] = (_Float16)w1[(size_t)l*CC*FFD + (size_t)k*FFD + n];
  }
  for (int i = t0; i < n1; i += st) {
    const int l = i / (CC*FFD), rem = i % (CC*FFD), n = rem / FFD, k = rem % FFD;
    w2t[i] = (_Float16)w2[(size_t)l*FFD*CC + (size_t)k*CC + n];
  }
  const int n2 = LYR*PP*CC;
  for (int i = t0; i < n2; i += st) {
    const int l = i / (PP*CC), rem = i % (PP*CC), n = rem / CC, k = rem % CC;
    pwt[i] = (_Float16)pw[(size_t)l*CC*PP + (size_t)k*PP + n];
  }
  const int n3 = CC*3*PP;
  for (int i = t0; i < n3; i += st) {
    const int n = i / (3*PP), k = i % (3*PP);
    poswt[i] = (_Float16)posw[(size_t)k*CC + n];
  }
  const int n4 = CC*PP;
  for (int i = t0; i < n4; i += st) {
    const int n = i / PP, k = i % PP;
    gwt[i]  = (_Float16)gw[(size_t)k*CC + n];
    bewt[i] = (_Float16)bew[(size_t)k*CC + n];
  }
}

// ---------------- mutable state init ----------------------------------------
__global__ void init_state(const float* query, const float* qpos, const float* probs,
                           const float* refpts, float* qbuf, float* qposbuf,
                           float* logitsbuf, float* refs0) {
  const int t0 = blockIdx.x * blockDim.x + threadIdx.x;
  const int st = gridDim.x * blockDim.x;
  const int nQC = MTOT * CC;
  for (int i = t0; i < nQC; i += st) { qbuf[i] = query[i]; qposbuf[i] = qpos[i]; }
  const int nQP = MTOT * PP;
  for (int i = t0; i < nQP; i += st)
    logitsbuf[i] = logf(fmaxf(probs[i], 1e-8f));
  const int nR = MTOT * 3;
  for (int i = t0; i < nR; i += st) refs0[i] = refpts[i];
}

// ---------------- pos branch: ln(coords_flat @ pos_w + pos_b), hoisted ------
__global__ __launch_bounds__(256)
void pos_kernel(const float* coords, const _Float16* poswt, const float* posb,
                float* poslnbuf) {
  extern __shared__ char smem[];
  float*    rsum   = (float*)smem;                 // 64
  float*    rsumsq = rsum + 64;                    // 64
  _Float16* Ach    = (_Float16*)(rsumsq + 64);     // 64 x 40  (A chunk, row-major)
  _Float16* Wps    = Ach + 64*40;                  // 256 x 40 (B^T chunk [n][k])

  const int tid = threadIdx.x, lane = tid & 31, wave = tid >> 5;
  const int rowBase = blockIdx.x * BM;
  const int mt = wave & 3, nb = (wave >> 2) * 8;
  const int rb = (lane >> 4) << 3;

  if (tid < 64) { rsum[tid] = 0.f; rsumsq[tid] = 0.f; }
  v8f acc[8];
#pragma unroll
  for (int t = 0; t < 8; ++t) acc[t] = (v8f)VZERO8;

  for (int kk = 0; kk < 3*PP; kk += 32) {
#ifdef HAVE_TDM
    if (wave == 0)
      tdm_load_2d(lds_addr_of(Wps), poswt + kk, 256, 32, 3*PP, 3u, 3u);
#else
    for (int idx = tid; idx < 256*4; idx += 256) {
      const int n = idx >> 2, kv = (idx & 3) << 3;
      *(v8h*)(Wps + n*40 + kv) = *(const v8h*)(poswt + (size_t)n*(3*PP) + kk + kv);
    }
#endif
    for (int idx = tid; idx < 64*8; idx += 256) {          // A chunk, f32->f16
      const int r = idx >> 3, c = (idx & 7) << 2;
      const float4 f = *(const float4*)(coords + (size_t)(rowBase + r)*(3*PP) + kk + c);
      v4h h; h[0]=(_Float16)f.x; h[1]=(_Float16)f.y; h[2]=(_Float16)f.z; h[3]=(_Float16)f.w;
      *(v4h*)(Ach + r*40 + c) = h;
    }
#ifdef HAVE_TDM
    if (wave == 0) TDM_WAIT();
#endif
    __syncthreads();
    const v16h a = frag_a(Ach + mt*16*40, 40, lane);
#pragma unroll
    for (int t = 0; t < 8; ++t)
      acc[t] = wmma32(a, frag_bt(Wps + (nb + t)*16*40, 40, lane), acc[t]);
    __syncthreads();
  }

#pragma unroll
  for (int t = 0; t < 8; ++t) {
    const int col = (nb + t)*16 + (lane & 15);
#pragma unroll
    for (int r = 0; r < 8; ++r) acc[t][r] += posb[col];
  }
#pragma unroll
  for (int r = 0; r < 8; ++r) {
    const int lrow = mt*16 + rb + r;
    float s = 0.f, s2 = 0.f;
#pragma unroll
    for (int t = 0; t < 8; ++t) { const float v = acc[t][r]; s += v; s2 += v*v; }
    atomicAdd(&rsum[lrow], s);
    atomicAdd(&rsumsq[lrow], s2);
  }
  __syncthreads();
#pragma unroll
  for (int r = 0; r < 8; ++r) {
    const int lrow = mt*16 + rb + r;
    const float mean = rsum[lrow] * (1.f/CC);
    const float var  = rsumsq[lrow] * (1.f/CC) - mean*mean;
    const float rstd = rsqrtf(var + 1e-5f);
#pragma unroll
    for (int t = 0; t < 8; ++t) {
      const int col = (nb + t)*16 + (lane & 15);
      poslnbuf[(size_t)(rowBase + lrow)*CC + col] = (acc[t][r] - mean) * rstd;
    }
  }
}

// ---------------- residual FFN + post-norm (interm) --------------------------
__global__ __launch_bounds__(256)
void ffn_kernel(float* qbuf, const float* qposbuf,
                const _Float16* w1t, const float* b1,
                const _Float16* w2t, const float* b2,
                const float* pnsc, const float* pnbi, float* interm_out) {
  extern __shared__ char smem[];
  float*    rsum   = (float*)smem;                 // 64
  float*    rsumsq = rsum + 64;                    // 64
  _Float16* Xs   = (_Float16*)(rsumsq + 64);       // 64 x 264  (A, row-major)
  _Float16* W1ts = Xs + 64*264;                    // 32 x 264  (B^T chunk [n][k=C])
  _Float16* Hs   = W1ts + 32*264;                  // 64 x 40   (A, row-major)
  _Float16* W2ts = Hs + 64*40;                     // 256 x 40  (B^T chunk [n][k=32])

  const int tid = threadIdx.x, lane = tid & 31, wave = tid >> 5;
  const int rowBase = blockIdx.x * BM;
  const int mt = wave & 3;
  const int g1nt = wave >> 2;
  const int g2nb = (wave >> 2) * 8;
  const int rb = (lane >> 4) << 3;

  for (int idx = tid; idx < BM*CC/4; idx += 256) {
    const int r = idx >> 6, c = (idx & 63) << 2;
    const size_t g = (size_t)(rowBase + r)*CC + c;
    const float4 qa = *(const float4*)(qbuf + g);
    const float4 qb = *(const float4*)(qposbuf + g);
    v4h h;
    h[0]=(_Float16)(qa.x+qb.x); h[1]=(_Float16)(qa.y+qb.y);
    h[2]=(_Float16)(qa.z+qb.z); h[3]=(_Float16)(qa.w+qb.w);
    *(v4h*)(Xs + r*264 + c) = h;
  }
  if (tid < 64) { rsum[tid] = 0.f; rsumsq[tid] = 0.f; }
  __syncthreads();

  v8f cacc[8];
#pragma unroll
  for (int t = 0; t < 8; ++t) cacc[t] = (v8f)VZERO8;

  for (int n0 = 0; n0 < FFD; n0 += 32) {
#ifdef HAVE_TDM
    if (wave == 0) {
      tdm_load_2d(lds_addr_of(W1ts), w1t + (size_t)n0*CC, 32, 256, CC, 6u, 3u);
      tdm_load_2d(lds_addr_of(W2ts), w2t + n0,           256, 32, FFD, 3u, 3u);
      TDM_WAIT();
    }
#else
    for (int idx = tid; idx < 32*32; idx += 256) {
      const int n = idx >> 5, kv = (idx & 31) << 3;
      *(v8h*)(W1ts + n*264 + kv) = *(const v8h*)(w1t + (size_t)(n0 + n)*CC + kv);
    }
    for (int idx = tid; idx < 256*4; idx += 256) {
      const int n = idx >> 2, kv = (idx & 3) << 3;
      *(v8h*)(W2ts + n*40 + kv) = *(const v8h*)(w2t + (size_t)n*FFD + n0 + kv);
    }
#endif
    __syncthreads();

    // GEMM1: H[64,32] = relu(X[64,256] @ W1chunk + b1)
    v8f h = (v8f)VZERO8;
#pragma unroll
    for (int kk = 0; kk < CC; kk += 32) {
      const v16h a   = frag_a(Xs + mt*16*264 + kk, 264, lane);
      const v16h bfr = frag_bt(W1ts + (g1nt*16)*264 + kk, 264, lane);
      h = wmma32(a, bfr, h);
    }
    {
      const int nloc = g1nt*16 + (lane & 15);
      const float bias1 = b1[n0 + nloc];
#pragma unroll
      for (int r = 0; r < 8; ++r) {
        float v = h[r] + bias1;
        v = v > 0.f ? v : 0.f;
        Hs[(mt*16 + rb + r)*40 + nloc] = (_Float16)v;
      }
    }
    __syncthreads();

    // GEMM2 partial: out += H[64,32] @ W2chunk[32,256]
    const v16h a2 = frag_a(Hs + mt*16*40, 40, lane);
#pragma unroll
    for (int t = 0; t < 8; ++t) {
      const v16h bfr = frag_bt(W2ts + ((g2nb + t)*16)*40, 40, lane);
      cacc[t] = wmma32(a2, bfr, cacc[t]);
    }
    __syncthreads();
  }

  // epilogue: residual add, write new query, post-norm (interm)
#pragma unroll
  for (int t = 0; t < 8; ++t) {
    const int col = (g2nb + t)*16 + (lane & 15);
#pragma unroll
    for (int r = 0; r < 8; ++r) {
      const int lrow = mt*16 + rb + r;
      const size_t g = (size_t)(rowBase + lrow)*CC + col;
      const float v = cacc[t][r] + b2[col] + qbuf[g];
      qbuf[g] = v;
      cacc[t][r] = v;
    }
  }
#pragma unroll
  for (int r = 0; r < 8; ++r) {
    const int lrow = mt*16 + rb + r;
    float s = 0.f, s2 = 0.f;
#pragma unroll
    for (int t = 0; t < 8; ++t) { const float v = cacc[t][r]; s += v; s2 += v*v; }
    atomicAdd(&rsum[lrow], s);
    atomicAdd(&rsumsq[lrow], s2);
  }
  __syncthreads();
#pragma unroll
  for (int r = 0; r < 8; ++r) {
    const int lrow = mt*16 + rb + r;
    const float mean = rsum[lrow] * (1.f/CC);
    const float var  = rsumsq[lrow] * (1.f/CC) - mean*mean;
    const float rstd = rsqrtf(var + 1e-5f);
#pragma unroll
    for (int t = 0; t < 8; ++t) {
      const int col = (g2nb + t)*16 + (lane & 15);
      interm_out[(size_t)(rowBase + lrow)*CC + col] =
          (cacc[t][r] - mean) * rstd * pnsc[col] + pnbi[col];
    }
  }
}

// ---------------- dyn-query: logits, softmax, refs, query_pos ----------------
__global__ __launch_bounds__(256)
void dyn_kernel(const float* qbuf, const _Float16* pwt, const float* pb,
                const unsigned char* mask, const float* coords, const float* ref0,
                const _Float16* gwt, const float* gb,
                const _Float16* bewt, const float* beb,
                const float* poslnbuf, float* logitsbuf, float* qposbuf,
                float* logits_out, float* refs_out) {
  extern __shared__ char smem[];
  float*    Ls = (float*)smem;               // 64 x 136 (logits -> probs, f32)
  _Float16* Qs = (_Float16*)(Ls + 64*136);   // 64 x 264 (A, row-major)
  _Float16* Ph = Qs + 64*264;                // 64 x 136 (probs f16, A)
  _Float16* Wg = Ph + 64*136;                // up to 256 x 40 (B^T chunks)

  const int tid = threadIdx.x, lane = tid & 31, wave = tid >> 5;
  const int rowBase = blockIdx.x * BM;
  const int rb = (lane >> 4) << 3;

  for (int idx = tid; idx < BM*CC/4; idx += 256) {
    const int r = idx >> 6, c = (idx & 63) << 2;
    const float4 f = *(const float4*)(qbuf + (size_t)(rowBase + r)*CC + c);
    v4h h; h[0]=(_Float16)f.x; h[1]=(_Float16)f.y; h[2]=(_Float16)f.z; h[3]=(_Float16)f.w;
    *(v4h*)(Qs + r*264 + c) = h;
  }
  __syncthreads();

  // GEMM: logits_delta = query @ prob_w  (M=64, K=256, N=128)
  {
    const int mt = wave >> 1, nh = wave & 1;
    v8f pacc[4];
#pragma unroll
    for (int t = 0; t < 4; ++t) pacc[t] = (v8f)VZERO8;
    for (int kk = 0; kk < CC; kk += 32) {
#ifdef HAVE_TDM
      if (wave == 0) {
        tdm_load_2d(lds_addr_of(Wg), pwt + kk, 128, 32, CC, 3u, 3u);
        TDM_WAIT();
      }
#else
      for (int idx = tid; idx < 128*4; idx += 256) {
        const int n = idx >> 2, kv = (idx & 3) << 3;
        *(v8h*)(Wg + n*40 + kv) = *(const v8h*)(pwt + (size_t)n*CC + kk + kv);
      }
#endif
      __syncthreads();
      const v16h a = frag_a(Qs + mt*16*264 + kk, 264, lane);
#pragma unroll
      for (int t = 0; t < 4; ++t)
        pacc[t] = wmma32(a, frag_bt(Wg + ((nh*4 + t)*16)*40, 40, lane), pacc[t]);
      __syncthreads();
    }
#pragma unroll
    for (int t = 0; t < 4; ++t) {
      const int col = (nh*4 + t)*16 + (lane & 15);
#pragma unroll
      for (int r = 0; r < 8; ++r) {
        const int lrow = mt*16 + rb + r;
        const int grow = rowBase + lrow;
        const float mf = mask[grow] ? 1.f : 0.f;
        const float lg = logitsbuf[(size_t)grow*PP + col] + (pacc[t][r] + pb[col]) * mf;
        logitsbuf[(size_t)grow*PP + col] = lg;
        logits_out[(size_t)grow*PP + col] = lg;
        Ls[lrow*136 + col] = lg;
      }
    }
  }
  __syncthreads();

  // softmax over P + probs.coords reference update (one thread per row)
  if (tid < 64) {
    const int lrow = tid, grow = rowBase + tid;
    float mx = -3.4e38f;
    for (int p = 0; p < PP; ++p) mx = fmaxf(mx, Ls[lrow*136 + p]);
    float s = 0.f;
    for (int p = 0; p < PP; ++p) {
      const float e = expf(Ls[lrow*136 + p] - mx);
      Ls[lrow*136 + p] = e; s += e;
    }
    const float inv = 1.f / s;
    float r0 = 0.f, r1 = 0.f, r2 = 0.f;
    const float* cr = coords + (size_t)grow*(3*PP);
    for (int p = 0; p < PP; ++p) {
      const float pr = Ls[lrow*136 + p] * inv;
      Ls[lrow*136 + p] = pr;
      Ph[lrow*136 + p] = (_Float16)pr;
      r0 += pr * cr[p*3 + 0];
      r1 += pr * cr[p*3 + 1];
      r2 += pr * cr[p*3 + 2];
    }
    const bool m = mask[grow] != 0;
    refs_out[(size_t)grow*3 + 0] = m ? r0 : ref0[(size_t)grow*3 + 0];
    refs_out[(size_t)grow*3 + 1] = m ? r1 : ref0[(size_t)grow*3 + 1];
    refs_out[(size_t)grow*3 + 2] = m ? r2 : ref0[(size_t)grow*3 + 2];
  }
  __syncthreads();

  // GEMMs: gain = probs @ g_w, beta = probs @ be_w (M=64, K=128, N=256)
  const int mt = wave & 3, nb = (wave >> 2) * 8;
  v8f gacc[8], beacc[8];
#pragma unroll
  for (int t = 0; t < 8; ++t) { gacc[t] = (v8f)VZERO8; beacc[t] = (v8f)VZERO8; }

  for (int kk = 0; kk < PP; kk += 32) {
#ifdef HAVE_TDM
    if (wave == 0) {
      tdm_load_2d(lds_addr_of(Wg), gwt + kk, 256, 32, PP, 3u, 3u);
      TDM_WAIT();
    }
#else
    for (int idx = tid; idx < 256*4; idx += 256) {
      const int n = idx >> 2, kv = (idx & 3) << 3;
      *(v8h*)(Wg + n*40 + kv) = *(const v8h*)(gwt + (size_t)n*PP + kk + kv);
    }
#endif
    __syncthreads();
    const v16h a = frag_a(Ph + mt*16*136 + kk, 136, lane);
#pragma unroll
    for (int t = 0; t < 8; ++t)
      gacc[t] = wmma32(a, frag_bt(Wg + ((nb + t)*16)*40, 40, lane), gacc[t]);
    __syncthreads();
  }
  for (int kk = 0; kk < PP; kk += 32) {
#ifdef HAVE_TDM
    if (wave == 0) {
      tdm_load_2d(lds_addr_of(Wg), bewt + kk, 256, 32, PP, 3u, 3u);
      TDM_WAIT();
    }
#else
    for (int idx = tid; idx < 256*4; idx += 256) {
      const int n = idx >> 2, kv = (idx & 3) << 3;
      *(v8h*)(Wg + n*40 + kv) = *(const v8h*)(bewt + (size_t)n*PP + kk + kv);
    }
#endif
    __syncthreads();
    const v16h a = frag_a(Ph + mt*16*136 + kk, 136, lane);
#pragma unroll
    for (int t = 0; t < 8; ++t)
      beacc[t] = wmma32(a, frag_bt(Wg + ((nb + t)*16)*40, 40, lane), beacc[t]);
    __syncthreads();
  }

  // masked query_pos update: ln(pos_base) * gain + beta
#pragma unroll
  for (int t = 0; t < 8; ++t) {
    const int col = (nb + t)*16 + (lane & 15);
#pragma unroll
    for (int r = 0; r < 8; ++r) {
      const int lrow = mt*16 + rb + r;
      const int grow = rowBase + lrow;
      if (mask[grow]) {
        const float gv = gacc[t][r] + gb[col];
        const float bv = beacc[t][r] + beb[col];
        qposbuf[(size_t)grow*CC + col] = poslnbuf[(size_t)grow*CC + col] * gv + bv;
      }
    }
  }
}

// ---------------- host orchestration ----------------------------------------
extern "C" void kernel_launch(void* const* d_in, const int* in_sizes, int n_in,
                              void* d_out, int out_size, void* d_ws, size_t ws_size,
                              hipStream_t stream) {
  const float* query     = (const float*)d_in[0];
  const float* query_pos = (const float*)d_in[1];
  const float* refpts    = (const float*)d_in[2];
  const float* coords    = (const float*)d_in[3];
  const float* probs     = (const float*)d_in[4];
  const unsigned char* mask = (const unsigned char*)d_in[5];
  const float* w1   = (const float*)d_in[6];
  const float* b1   = (const float*)d_in[7];
  const float* w2   = (const float*)d_in[8];
  const float* b2   = (const float*)d_in[9];
  const float* pw   = (const float*)d_in[10];
  const float* pb   = (const float*)d_in[11];
  const float* posw = (const float*)d_in[12];
  const float* posb = (const float*)d_in[13];
  const float* gw   = (const float*)d_in[14];
  const float* gb   = (const float*)d_in[15];
  const float* bew  = (const float*)d_in[16];
  const float* beb  = (const float*)d_in[17];
  const float* pnsc = (const float*)d_in[18];
  const float* pnbi = (const float*)d_in[19];

  float* wsf = (float*)d_ws;
  float* qbuf      = wsf;
  float* qposbuf   = qbuf      + (size_t)MTOT*CC;
  float* logitsbuf = qposbuf   + (size_t)MTOT*CC;
  float* poslnbuf  = logitsbuf + (size_t)MTOT*PP;
  _Float16* w1t   = (_Float16*)(poslnbuf + (size_t)MTOT*CC);
  _Float16* w2t   = w1t   + (size_t)LYR*FFD*CC;
  _Float16* pwt   = w2t   + (size_t)LYR*CC*FFD;
  _Float16* poswt = pwt   + (size_t)LYR*PP*CC;
  _Float16* gwt   = poswt + (size_t)CC*3*PP;
  _Float16* bewt  = gwt   + (size_t)CC*PP;

  float* out        = (float*)d_out;
  float* interm_out = out;
  float* refs_out   = out + (size_t)LYR*MTOT*CC;
  float* logits_out = refs_out + (size_t)(LYR+1)*MTOT*3;

  prep_weights<<<512, 256, 0, stream>>>(w1, w2, pw, posw, gw, bew,
                                        w1t, w2t, pwt, poswt, gwt, bewt);
  init_state<<<1024, 256, 0, stream>>>(query, query_pos, probs, refpts,
                                       qbuf, qposbuf, logitsbuf, refs_out);

  const size_t smemPos = 128*4 + (64*40 + 256*40)*2;
  const size_t smemF   = 128*4 + (64*264 + 32*264 + 64*40 + 256*40)*2;
  const size_t smemD   = 64*136*4 + (64*264 + 64*136 + 256*40)*2;

  pos_kernel<<<MTOT/BM, 256, smemPos, stream>>>(coords, poswt, posb, poslnbuf);

  for (int l = 0; l < LYR; ++l) {
    ffn_kernel<<<MTOT/BM, 256, smemF, stream>>>(
        qbuf, qposbuf,
        w1t + (size_t)l*FFD*CC, b1 + (size_t)l*FFD,
        w2t + (size_t)l*CC*FFD, b2 + (size_t)l*CC,
        pnsc, pnbi, interm_out + (size_t)l*MTOT*CC);
    dyn_kernel<<<MTOT/BM, 256, smemD, stream>>>(
        qbuf, pwt + (size_t)l*PP*CC, pb + (size_t)l*PP, mask,
        coords, refpts, gwt, gb, bewt, beb, poslnbuf,
        logitsbuf, qposbuf,
        logits_out + (size_t)l*MTOT*PP, refs_out + (size_t)(l+1)*MTOT*3);
  }
}
